// KNNTorch_75419625717855
// MI455X (gfx1250) — compile-verified, matrix operand order checked by reading, not run on hardware
//
#include <hip/hip_runtime.h>
#include <stdint.h>
#include <stddef.h>

// ---------------- problem constants ----------------
#define B_ROWS 4096
#define N_REF  50000
#define K_DIM  256
#define K_NN   10
#define N_CHUNKS 8
#define CHUNK_COLS (N_REF / N_CHUNKS)              // 6250
#define TILES_PER_CHUNK ((CHUNK_COLS + 15) / 16)   // 391 (last tile: 10 valid cols)
#define ROWS_PER_BLOCK 128                          // 8 waves * 16 rows
#define PARTIAL_STRIDE (N_CHUNKS * 2 * K_NN)        // 160 floats per row

typedef __bf16 bf16;
typedef __attribute__((ext_vector_type(16))) __bf16        v16bf;
typedef __attribute__((ext_vector_type(8)))  float         v8f;
typedef __attribute__((ext_vector_type(4)))  unsigned int  v4u;
typedef __attribute__((ext_vector_type(8)))  int           v8i;
typedef __attribute__((ext_vector_type(4)))  int           v4i;

// ---------------- small device helpers ----------------
__device__ __forceinline__ uint32_t lds_off(const void* p) {
  // flat LDS aperture: addr[31:0] is the LDS byte offset
  return (uint32_t)(uintptr_t)p;
}

__device__ __forceinline__ v16bf load2x8(const bf16* base, int k0, int k1) {
  union { v4u q[2]; v16bf v; } u;
  u.q[0] = *(const v4u*)(base + k0);   // K = k0..k0+7  (16B)
  u.q[1] = *(const v4u*)(base + k1);   // K = k1..k1+7  (16B)
  return u.v;
}

__device__ __forceinline__ v8f wmma_bf16(v16bf a, v16bf b, v8f c) {
  return __builtin_amdgcn_wmma_f32_16x16x32_bf16(
      /*neg_a=*/false, a, /*neg_b=*/false, b,
      /*c_mod=*/(short)0, c, /*reuse_a=*/false, /*reuse_b=*/false);
}

__device__ __forceinline__ void topk_insert(float (&hp)[K_NN], float v) {
  if (v < hp[K_NN - 1]) {
    hp[K_NN - 1] = v;
#pragma unroll
    for (int q = K_NN - 1; q > 0; --q) {      // one bubble pass restores sorted order
      float a = fminf(hp[q - 1], hp[q]);
      float b = fmaxf(hp[q - 1], hp[q]);
      hp[q - 1] = a; hp[q] = b;
    }
  }
}

// TDM: 2D tile load (bf16 elements, data_size=2B) of [tile1=16 rows] x [tile0=256 elems]
// from a row-major [dim1 x 256] tensor starting at gbase row n0, into LDS at ldsOff.
// tensor_dim1 = rows remaining => OOB rows of the last partial tile read back zero.
__device__ __forceinline__ void tdm_tile_load(const bf16* gbase, int n0, uint32_t ldsOff) {
  uint32_t dim1 = (uint32_t)(N_REF - n0);
  uint64_t ga = (uint64_t)(uintptr_t)(gbase + (size_t)n0 * K_DIM);
  v4u g0;
  g0.x = 1u;                                             // count=1 (valid user descriptor)
  g0.y = ldsOff;                                         // lds_addr
  g0.z = (uint32_t)ga;                                   // global_addr[31:0]
  g0.w = (uint32_t)((ga >> 32) & 0x1FFFFFFu) | 0x80000000u; // addr[56:32] | type=2
  v8i g1;
  g1[0] = (int)(1u << 16);                               // wg_mask=0, data_size=1 (2 bytes)
  g1[1] = (int)((K_DIM & 0xFFFF) << 16);                 // tensor_dim0 lo16 (=256)
  g1[2] = (int)(((K_DIM >> 16) & 0xFFFF) | ((dim1 & 0xFFFFu) << 16)); // dim0 hi | dim1 lo
  g1[3] = (int)(((dim1 >> 16) & 0xFFFFu) | ((K_DIM & 0xFFFF) << 16)); // dim1 hi | tile_dim0=256
  g1[4] = 16;                                            // tile_dim1=16, tile_dim2=0 (2D)
  g1[5] = K_DIM;                                         // tensor_dim0_stride lo32 (=256 elems)
  g1[6] = 0;
  g1[7] = 0;
  v4i z4 = {0, 0, 0, 0};
#if defined(__clang_major__) && (__clang_major__ >= 23)
  v8i z8 = {0, 0, 0, 0, 0, 0, 0, 0};
  __builtin_amdgcn_tensor_load_to_lds(g0, g1, z4, z4, z8, 0);
#else
  __builtin_amdgcn_tensor_load_to_lds(g0, g1, z4, z4, 0);
#endif
}

// ---------------- kernel 1: f32 -> bf16 hi/lo split + row norms ----------------
// One 256-thread block per row; first B_ROWS blocks handle X, rest handle x_ref.
__global__ __launch_bounds__(256) void knn_prep(
    const float* __restrict__ X, const float* __restrict__ xref,
    bf16* __restrict__ Xhi, bf16* __restrict__ Xlo,
    bf16* __restrict__ Rhi, bf16* __restrict__ Rlo,
    float* __restrict__ x2, float* __restrict__ y2) {
  const int r = blockIdx.x;
  const int t = threadIdx.x;
  const float* src;
  bf16 *dhi, *dlo;
  float* nrm;
  if (r < B_ROWS) {
    src = X + (size_t)r * K_DIM;
    dhi = Xhi + (size_t)r * K_DIM;
    dlo = Xlo + (size_t)r * K_DIM;
    nrm = x2 + r;
  } else {
    const int rr = r - B_ROWS;
    src = xref + (size_t)rr * K_DIM;
    dhi = Rhi + (size_t)rr * K_DIM;
    dlo = Rlo + (size_t)rr * K_DIM;
    nrm = y2 + rr;
  }
  const float v = src[t];
  const bf16 h = (bf16)v;
  const float hf = (float)h;
  dhi[t] = h;
  dlo[t] = (bf16)(v - hf);

  __shared__ float red[256];
  red[t] = v * v;
  __syncthreads();
#pragma unroll
  for (int s = 128; s > 0; s >>= 1) {
    if (t < s) red[t] += red[t + s];
    __syncthreads();
  }
  if (t == 0) *nrm = red[0];
}

// ---------------- kernel 2: WMMA distances + per-row partial top-k ----------------
// grid = (B_ROWS/128, N_CHUNKS), block = 256 threads (8 waves).
// Wave w owns rows [blockIdx.x*128 + w*16, +16). B tiles (16 ref cols, hi+lo bf16)
// are TDM-streamed into double-buffered LDS and shared by all 8 waves.
__global__ __launch_bounds__(256) void knn_main(
    const bf16* __restrict__ Xhi, const bf16* __restrict__ Xlo,
    const bf16* __restrict__ Rhi, const bf16* __restrict__ Rlo,
    const float* __restrict__ x2, const float* __restrict__ y2,
    float* __restrict__ partial) {
  __shared__ __align__(16) bf16 Bbuf[2][2][16][K_DIM];   // [buf][hi/lo][col][k] = 32 KB
  __shared__ float Sq[8][16][17];                        // padded transpose buffers

  const int tid  = threadIdx.x;
  const int wave = tid >> 5;
  const int lane = tid & 31;
  const int c    = lane & 15;   // A row (within tile) / B col / C column
  const int h    = lane >> 4;   // lane half: selects K sub-range and C row-half
  const int rowbase    = blockIdx.x * ROWS_PER_BLOCK + wave * 16;
  const int chunk      = blockIdx.y;
  const int chunkStart = chunk * CHUNK_COLS;
  const int chunkEnd   = chunkStart + CHUNK_COLS;
  const float FINF = __builtin_inff();

  // A fragments for this wave's 16 rows, all 8 K-steps, hi and lo (held in VGPRs)
  const bf16* arow_hi = Xhi + (size_t)(rowbase + c) * K_DIM;
  const bf16* arow_lo = Xlo + (size_t)(rowbase + c) * K_DIM;
  v16bf Ahi[8], Alo[8];
#pragma unroll
  for (int s = 0; s < 8; ++s) {
    const int k0 = 32 * s + 8 * h;
    Ahi[s] = load2x8(arow_hi, k0, k0 + 16);
    Alo[s] = load2x8(arow_lo, k0, k0 + 16);
  }
  float x2v[8];
#pragma unroll
  for (int j = 0; j < 8; ++j) x2v[j] = x2[rowbase + h * 8 + j];

  float hp[K_NN];
#pragma unroll
  for (int j = 0; j < K_NN; ++j) hp[j] = FINF;

  // preload tile 0 (wave 0 only; barrier publishes it to all waves)
  if (tid < 32) {
    tdm_tile_load(Rhi, chunkStart, lds_off(&Bbuf[0][0][0][0]));
    tdm_tile_load(Rlo, chunkStart, lds_off(&Bbuf[0][1][0][0]));
  }
  __builtin_amdgcn_s_wait_tensorcnt(0);
  __syncthreads();

  int p = 0;
  for (int i = 0; i < TILES_PER_CHUNK; ++i) {
    const int n0 = chunkStart + i * 16;
    // async-prefetch next B tile into the other buffer via the TDM
    if ((i + 1 < TILES_PER_CHUNK) && (tid < 32)) {
      const int n1 = chunkStart + (i + 1) * 16;
      tdm_tile_load(Rhi, n1, lds_off(&Bbuf[p ^ 1][0][0][0]));
      tdm_tile_load(Rlo, n1, lds_off(&Bbuf[p ^ 1][1][0][0]));
    }

    // 16x16 dot-product tile over K=256, hi/lo bf16 split (hi*hi + lo*hi + hi*lo).
    // Two explicit passes over the B tile keep only 2 B fragments live at a time
    // (current + prefetched), so loads stay ahead of the WMMAs; independent
    // accumulator chains keep same-accumulator RAW distance at 2.
    const bf16* bbase_h = &Bbuf[p][0][c][0];
    const bf16* bbase_l = &Bbuf[p][1][c][0];
    v8f acc_hh = {}, acc_lh = {}, acc_hl0 = {}, acc_hl1 = {};

    {  // pass A: hi-B fragments, 2 WMMAs each (alternating chains)
      v16bf bh = load2x8(bbase_h, 8 * h, 8 * h + 16);
#pragma unroll
      for (int s = 0; s < 8; ++s) {
        v16bf bh_n;
        if (s < 7) {
          const int kn = 32 * (s + 1) + 8 * h;
          bh_n = load2x8(bbase_h, kn, kn + 16);
        }
        acc_hh = wmma_bf16(Ahi[s], bh, acc_hh);
        acc_lh = wmma_bf16(Alo[s], bh, acc_lh);
        bh = bh_n;
      }
    }
    {  // pass B: lo-B fragments, 1 WMMA each (two alternating accumulators)
      v16bf bl = load2x8(bbase_l, 8 * h, 8 * h + 16);
#pragma unroll
      for (int s = 0; s < 8; ++s) {
        v16bf bl_n;
        if (s < 7) {
          const int kn = 32 * (s + 1) + 8 * h;
          bl_n = load2x8(bbase_l, kn, kn + 16);
        }
        if (s & 1) acc_hl1 = wmma_bf16(Ahi[s], bl, acc_hl1);
        else       acc_hl0 = wmma_bf16(Ahi[s], bl, acc_hl0);
        bl = bl_n;
      }
    }

    // squared distance epilogue; write transposed through LDS
    {
      const int cc = n0 + c;
      const float y2c = y2[cc < (N_REF - 1) ? cc : (N_REF - 1)];
#pragma unroll
      for (int j = 0; j < 8; ++j) {
        const float dot = (acc_hh[j] + acc_lh[j]) + (acc_hl0[j] + acc_hl1[j]);
        Sq[wave][h * 8 + j][c] = x2v[j] + y2c - 2.0f * dot;
      }
    }

    __builtin_amdgcn_s_wait_tensorcnt(0);  // next-tile TDM done (no-op for waves 1..7)
    __syncthreads();                       // publishes Sq lanes + Bbuf[p^1]; retires Bbuf[p] reads

    // lane owns row (lane&15); consume its 8 columns of this tile
#pragma unroll
    for (int j = 0; j < 8; ++j) {
      const int col = n0 + h * 8 + j;
      const float v = (col < chunkEnd) ? Sq[wave][c][h * 8 + j] : FINF;
      topk_insert(hp, v);
    }
    p ^= 1;
  }

  // dump this lane's partial top-k list (both halves kept; merged in finalize)
  float* outp = partial + (size_t)(rowbase + c) * PARTIAL_STRIDE + chunk * (2 * K_NN) + h * K_NN;
#pragma unroll
  for (int j = 0; j < K_NN; ++j) outp[j] = hp[j];
}

// ---------------- kernel 3: merge partials, emit sqrt of k-th smallest ----------------
__global__ __launch_bounds__(256) void knn_finalize(
    const float* __restrict__ partial, float* __restrict__ out) {
  const int r = blockIdx.x * blockDim.x + threadIdx.x;
  if (r >= B_ROWS) return;
  const float FINF = __builtin_inff();
  float hp[K_NN];
#pragma unroll
  for (int j = 0; j < K_NN; ++j) hp[j] = FINF;
  const float* p = partial + (size_t)r * PARTIAL_STRIDE;
  for (int i = 0; i < PARTIAL_STRIDE; ++i) topk_insert(hp, p[i]);
  out[r] = sqrtf(fmaxf(hp[K_NN - 1], 0.0f));
}

// ---------------- host launch ----------------
extern "C" void kernel_launch(void* const* d_in, const int* in_sizes, int n_in,
                              void* d_out, int out_size, void* d_ws, size_t ws_size,
                              hipStream_t stream) {
  (void)in_sizes; (void)n_in; (void)out_size; (void)ws_size;
  const float* X    = (const float*)d_in[0];   // [4096, 256] f32
  const float* xref = (const float*)d_in[1];   // [50000, 256] f32
  float* out = (float*)d_out;                  // [4096] f32

  char* base = (char*)d_ws;
  size_t off = 0;
  auto take = [&](size_t bytes) {
    char* q = base + off;
    off += (bytes + 255) & ~(size_t)255;
    return (void*)q;
  };
  bf16* Rhi = (bf16*)take((size_t)N_REF * K_DIM * sizeof(bf16));    // 25.6 MB
  bf16* Rlo = (bf16*)take((size_t)N_REF * K_DIM * sizeof(bf16));    // 25.6 MB
  bf16* Xhi = (bf16*)take((size_t)B_ROWS * K_DIM * sizeof(bf16));   //  2.1 MB
  bf16* Xlo = (bf16*)take((size_t)B_ROWS * K_DIM * sizeof(bf16));   //  2.1 MB
  float* x2 = (float*)take((size_t)B_ROWS * sizeof(float));
  float* y2 = (float*)take((size_t)N_REF * sizeof(float));
  float* partial = (float*)take((size_t)B_ROWS * PARTIAL_STRIDE * sizeof(float)); // 2.6 MB

  knn_prep<<<B_ROWS + N_REF, 256, 0, stream>>>(X, xref, Xhi, Xlo, Rhi, Rlo, x2, y2);
  knn_main<<<dim3(B_ROWS / ROWS_PER_BLOCK, N_CHUNKS), 256, 0, stream>>>(
      Xhi, Xlo, Rhi, Rlo, x2, y2, partial);
  knn_finalize<<<(B_ROWS + 255) / 256, 256, 0, stream>>>(partial, out);
}
